// RoPEAttention_40570261078178
// MI455X (gfx1250) — matrix operand order repaired and checked by reading, not compiled, not run on hardware
//
#include <hip/hip_runtime.h>

typedef __attribute__((ext_vector_type(16))) _Float16 v16h;
typedef __attribute__((ext_vector_type(8)))  _Float16 v8h;
typedef __attribute__((ext_vector_type(4)))  _Float16 v4h;
typedef __attribute__((ext_vector_type(2)))  _Float16 v2h;
typedef __attribute__((ext_vector_type(8)))  float    v8f;
typedef __attribute__((ext_vector_type(4)))  float    v4f;
typedef __attribute__((ext_vector_type(2)))  float    v2f;

// ---------------------------------------------------------------------------
// WMMA fragment helpers (CDNA5 V_WMMA_F32_16X16X32_F16 layouts, wave32)
// A: 16x32 f16. lane L -> row L%16; half i -> K = (i/8)*16 + (L/16)*8 + (i%8)
// B: 32x16 f16. lane L -> col L%16; half i -> K = (i/8)*16 + (L/16)*8 + (i%8)
// C: 16x16 f32, 8 VGPRs. vgpr r, lane L -> M = r + (L/16)*8, N = L%16
// ---------------------------------------------------------------------------
__device__ __forceinline__ v16h frag_a(const _Float16* __restrict__ smem, int ld,
                                       int m_base, int k_base) {
  const int lane = threadIdx.x & 31;
  const int m  = m_base + (lane & 15);
  const int kb = (lane >> 4) << 3;                 // 0 or 8
  const _Float16* p = smem + m * ld + k_base + kb;
  v8h lo = *(const v8h*)(p);
  v8h hi = *(const v8h*)(p + 16);
  v16h a;
#pragma unroll
  for (int i = 0; i < 8; ++i) { a[i] = lo[i]; a[i + 8] = hi[i]; }
  return a;
}

// B staged TRANSPOSED in LDS: smemT[n*ld + k] = B[k][n]
__device__ __forceinline__ v16h frag_b(const _Float16* __restrict__ smemT, int ld,
                                       int n_base, int k_base) {
  const int lane = threadIdx.x & 31;
  const int n  = n_base + (lane & 15);
  const int kk = k_base + ((lane >> 4) << 3);      // +0 or +8
  const _Float16* p = smemT + n * ld + kk;
  v8h lo = *(const v8h*)(p);
  v8h hi = *(const v8h*)(p + 16);
  v16h b;
#pragma unroll
  for (int i = 0; i < 8; ++i) { b[i] = lo[i]; b[i + 8] = hi[i]; }
  return b;
}

#define WMMA_F16(a, b, c) \
  __builtin_amdgcn_wmma_f32_16x16x32_f16(false, (a), false, (b), (short)0, (c), false, false)

// Per-lane 16B global -> LDS async DMA (gfx1250), tracked by ASYNCcnt.
__device__ __forceinline__ void async_b128(const void* gptr, const void* lptr) {
  asm volatile("global_load_async_to_lds_b128 %0, %1, off"
               :: "v"((unsigned int)(uintptr_t)lptr),
                  "v"((unsigned long long)(uintptr_t)gptr)
               : "memory");
}
__device__ __forceinline__ void wait_async0() {
  asm volatile("s_wait_asynccnt 0x0" ::: "memory");
}

// ---------------------------------------------------------------------------
// Generic GEMM + bias: Y[M,N] = f16(X[M,K]) @ f16(W[K,N]) + bias   (fp32 io)
// Block tile 64x128, K-step 32, 256 threads = 8 waves (4 x 2), 4 frags/wave.
// ---------------------------------------------------------------------------
__global__ __launch_bounds__(256) void gemm_bias_f16(
    const float* __restrict__ X, const float* __restrict__ W,
    const float* __restrict__ bias, float* __restrict__ Y,
    int Ndim, int Kdim) {
  constexpr int TM = 64, TN = 128, TK = 32, LDA = TK + 8, LDB = TK + 8;
  __shared__ _Float16 sA[TM * LDA];
  __shared__ _Float16 sBt[TN * LDB];
  const int tid = threadIdx.x;
  const int wid = tid >> 5;
  const int m0 = blockIdx.x * TM;
  const int n0 = blockIdx.y * TN;
  const int wm = (wid & 3) << 4;
  const int wn = (wid >> 2) << 6;
  v8f acc[4] = {};

  for (int kt = 0; kt < Kdim; kt += TK) {
    // A tile: 64x32 f32 -> f16 (512 float4 chunks, 2 per thread)
#pragma unroll
    for (int it = 0; it < 2; ++it) {
      int j = tid + it * 256;
      int r = j >> 3, c4 = (j & 7) << 2;
      v4f xv = *(const v4f*)(X + (size_t)(m0 + r) * Kdim + kt + c4);
      v4h hv;
#pragma unroll
      for (int q = 0; q < 4; ++q) hv[q] = (_Float16)xv[q];
      *(v4h*)(&sA[r * LDA + c4]) = hv;
    }
    // B tile: 32x128 f32 -> f16 transposed (1024 float4 chunks, 4 per thread)
#pragma unroll
    for (int it = 0; it < 4; ++it) {
      int j = tid + it * 256;
      int r = j >> 5, c4 = (j & 31) << 2;
      v4f wv = *(const v4f*)(W + (size_t)(kt + r) * Ndim + n0 + c4);
#pragma unroll
      for (int q = 0; q < 4; ++q) sBt[(c4 + q) * LDB + r] = (_Float16)wv[q];
    }
    __syncthreads();
    v16h a = frag_a(sA, LDA, wm, 0);
#pragma unroll
    for (int f = 0; f < 4; ++f) {
      v16h b = frag_b(sBt, LDB, wn + f * 16, 0);
      acc[f] = WMMA_F16(a, b, acc[f]);
    }
    __syncthreads();
  }
  const int lane = tid & 31;
  const int mrow = m0 + wm + ((lane >> 4) << 3);
  const int nc0  = n0 + wn + (lane & 15);
#pragma unroll
  for (int f = 0; f < 4; ++f) {
    int nc = nc0 + f * 16;
    float bv = bias[nc];
#pragma unroll
    for (int r = 0; r < 8; ++r)
      Y[(size_t)(mrow + r) * Ndim + nc] = acc[f][r] + bv;
  }
}

// ---------------------------------------------------------------------------
// RoPE: read rotated pair from f32 (b,s,h,dh), write f16 copy (optionally
// pre-scaled by 1/sqrt(Dh) for Q). One thread per (even,odd) pair.
// ---------------------------------------------------------------------------
__global__ __launch_bounds__(256) void rope_f16_kernel(
    const float* __restrict__ src, _Float16* __restrict__ dst, float scale) {
  const size_t idx = (size_t)blockIdx.x * 256 + threadIdx.x;  // 4096*512 pairs
  const int row  = (int)(idx >> 9);
  const int pair = (int)(idx & 511);
  const int h = pair >> 5, i = pair & 31;
  const int s = row & 2047;
  const float freq = __powf(10000.f, -(float)i * (1.f / 32.f));
  float sn, cs;
  __sincosf((float)s * freq, &sn, &cs);
  const size_t off = (size_t)row * 1024 + h * 64 + 2 * i;
  v2f eo = *(const v2f*)(src + off);
  v2h r;
  r[0] = (_Float16)((eo[0] * cs - eo[1] * sn) * scale);
  r[1] = (_Float16)((eo[0] * sn + eo[1] * cs) * scale);
  *(v2h*)(dst + off) = r;
}

// ---------------------------------------------------------------------------
// V transpose: f32 (b,s,h,d) -> f16 Vt[(bh*64+d)*2048 + s]  (contiguous in s)
// ---------------------------------------------------------------------------
__global__ __launch_bounds__(256) void vtrans_kernel(
    const float* __restrict__ V, _Float16* __restrict__ Vt) {
  const size_t idx = (size_t)blockIdx.x * 256 + threadIdx.x;  // 32*64*2048
  const int s = (int)(idx & 2047);
  const int t = (int)(idx >> 11);  // bh*64 + d
  const int d = t & 63, bh = t >> 6;
  const int b = bh >> 4, h = bh & 15;
  Vt[(size_t)t * 2048 + s] = (_Float16)V[((size_t)b * 2048 + s) * 1024 + h * 64 + d];
}

// ---------------------------------------------------------------------------
// Scores: per (b,h)  S = Qs @ K^T (Q pre-scaled), raw scores -> attn (f32).
// Q/K tiles staged f16 via global_load_async_to_lds_b128 (ASYNCcnt path).
// ---------------------------------------------------------------------------
__global__ __launch_bounds__(256) void scores_kernel(
    const _Float16* __restrict__ Q16, const _Float16* __restrict__ K16,
    float* __restrict__ Sout) {
  constexpr int TQ = 64, TKN = 128, LD = 72;   // 72 halves = 144B = 9*16B
  __shared__ _Float16 sQ[TQ * LD];
  __shared__ _Float16 sK[TKN * LD];
  const int tid = threadIdx.x, wid = tid >> 5, lane = tid & 31;
  const int bh = blockIdx.z, b = bh >> 4, h = bh & 15;
  const int q0 = blockIdx.x * TQ;
  const int n0 = blockIdx.y * TKN;
  const _Float16* Qh = Q16 + (size_t)b * 2048 * 1024 + h * 64;
  const _Float16* Kh = K16 + (size_t)b * 2048 * 1024 + h * 64;

  // Q tile: 64 rows x 64 halves = 512 x 16B chunks (2 rounds)
#pragma unroll
  for (int rnd = 0; rnd < 2; ++rnd) {
    int ch = rnd * 256 + tid;
    int r = ch >> 3, c = ch & 7;
    async_b128(Qh + (size_t)(q0 + r) * 1024 + c * 8, &sQ[r * LD + c * 8]);
  }
  // K tile: 128 rows x 64 halves = 1024 x 16B chunks (4 rounds)
#pragma unroll
  for (int rnd = 0; rnd < 4; ++rnd) {
    int ch = rnd * 256 + tid;
    int r = ch >> 3, c = ch & 7;
    async_b128(Kh + (size_t)(n0 + r) * 1024 + c * 8, &sK[r * LD + c * 8]);
  }
  wait_async0();
  __syncthreads();

  const int wm = (wid & 3) << 4, wn = (wid >> 2) << 6;
  v8f acc[4] = {};
#pragma unroll
  for (int k0 = 0; k0 < 64; k0 += 32) {
    v16h a = frag_a(sQ, LD, wm, k0);
#pragma unroll
    for (int f = 0; f < 4; ++f) {
      v16h bf = frag_b(sK, LD, wn + f * 16, k0);
      acc[f] = WMMA_F16(a, bf, acc[f]);
    }
  }
  const int mrow = q0 + wm + ((lane >> 4) << 3);
  const int nc0  = n0 + wn + (lane & 15);
  float* Sb = Sout + (size_t)bh * 2048 * 2048;
#pragma unroll
  for (int f = 0; f < 4; ++f)
#pragma unroll
    for (int r = 0; r < 8; ++r)
      Sb[(size_t)(mrow + r) * 2048 + nc0 + f * 16] = acc[f][r];
}

// ---------------------------------------------------------------------------
// Row softmax in place (2048 cols), values cached in registers.
// ---------------------------------------------------------------------------
__global__ __launch_bounds__(256) void softmax_kernel(float* __restrict__ P) {
  float* p = P + (size_t)blockIdx.x * 2048;
  const int tid = threadIdx.x;
  __shared__ float red_m[8];
  __shared__ float red_s[8];
  float v[8];
  float m = -3.402823466e38f;
#pragma unroll
  for (int i = 0; i < 8; ++i) { v[i] = p[tid + i * 256]; m = fmaxf(m, v[i]); }
#pragma unroll
  for (int off = 16; off > 0; off >>= 1) m = fmaxf(m, __shfl_xor(m, off, 32));
  if ((tid & 31) == 0) red_m[tid >> 5] = m;
  __syncthreads();
  m = red_m[0];
#pragma unroll
  for (int i = 1; i < 8; ++i) m = fmaxf(m, red_m[i]);
  float s = 0.f;
#pragma unroll
  for (int i = 0; i < 8; ++i) { v[i] = __expf(v[i] - m); s += v[i]; }
#pragma unroll
  for (int off = 16; off > 0; off >>= 1) s += __shfl_xor(s, off, 32);
  if ((tid & 31) == 0) red_s[tid >> 5] = s;
  __syncthreads();
  s = red_s[0];
#pragma unroll
  for (int i = 1; i < 8; ++i) s += red_s[i];
  const float inv = 1.f / s;
#pragma unroll
  for (int i = 0; i < 8; ++i) p[tid + i * 256] = v[i] * inv;
}

// ---------------------------------------------------------------------------
// O = P(2048x2048) @ V_h(2048x64) per (b,h). K-step 64: async DMA of the V^T
// tile is issued FIRST, overlapping with the f32->f16 conversion of the P
// tile; 4 WMMAs per barrier pair. 32 iterations.
// ---------------------------------------------------------------------------
__global__ __launch_bounds__(256) void attnv_kernel(
    const float* __restrict__ P, const _Float16* __restrict__ Vt,
    float* __restrict__ O) {
  constexpr int TM = 64, TK = 64, LDA = 72, LDB = 72;  // 144B rows = 9*16B
  __shared__ _Float16 sA[TM * LDA];
  __shared__ _Float16 sVt[64 * LDB];
  const int tid = threadIdx.x, wid = tid >> 5, lane = tid & 31;
  const int bh = blockIdx.z, b = bh >> 4, h = bh & 15;
  const int m0 = blockIdx.x * TM;
  const float*    Pb  = P  + (size_t)bh * 2048 * 2048;
  const _Float16* VtH = Vt + (size_t)bh * 64 * 2048;
  const int wm = (wid & 3) << 4, wn = (wid >> 2) << 5;  // wn = 0 or 32
  v8f acc[2] = {};

  for (int kt = 0; kt < 2048; kt += TK) {
    // 1) Vt tile 64 n-rows x 64 k-halves = 512 x 16B chunks: async DMA first
#pragma unroll
    for (int rnd = 0; rnd < 2; ++rnd) {
      int ch = rnd * 256 + tid;
      int n = ch >> 3, c = ch & 7;
      async_b128(VtH + (size_t)n * 2048 + kt + c * 8, &sVt[n * LDB + c * 8]);
    }
    // 2) A tile: 64x64 f32 -> f16 (1024 float4 chunks, 4/thread) overlaps DMA
#pragma unroll
    for (int it = 0; it < 4; ++it) {
      int j = tid + it * 256;
      int r = j >> 4, c4 = (j & 15) << 2;
      v4f pv = *(const v4f*)(Pb + (size_t)(m0 + r) * 2048 + kt + c4);
      v4h hv;
#pragma unroll
      for (int q = 0; q < 4; ++q) hv[q] = (_Float16)pv[q];
      *(v4h*)(&sA[r * LDA + c4]) = hv;
    }
    wait_async0();
    __syncthreads();
#pragma unroll
    for (int k0 = 0; k0 < TK; k0 += 32) {
      v16h a = frag_a(sA, LDA, wm, k0);
#pragma unroll
      for (int f = 0; f < 2; ++f) {
        v16h bf = frag_b(sVt, LDB, wn + f * 16, k0);
        acc[f] = WMMA_F16(a, bf, acc[f]);
      }
    }
    __syncthreads();
  }
  const int mrow = m0 + wm + ((lane >> 4) << 3);
  const int nc0  = wn + (lane & 15);
  float* Ob = O + (size_t)b * 2048 * 1024 + h * 64;
#pragma unroll
  for (int f = 0; f < 2; ++f)
#pragma unroll
    for (int r = 0; r < 8; ++r)
      Ob[(size_t)(mrow + r) * 1024 + nc0 + f * 16] = acc[f][r];
}

// ---------------------------------------------------------------------------
extern "C" void kernel_launch(void* const* d_in, const int* in_sizes, int n_in,
                              void* d_out, int out_size, void* d_ws, size_t ws_size,
                              hipStream_t stream) {
  const float* x  = (const float*)d_in[0];
  const float* wq = (const float*)d_in[1];
  const float* bq = (const float*)d_in[2];
  const float* wk = (const float*)d_in[3];
  const float* bk = (const float*)d_in[4];
  const float* wv = (const float*)d_in[5];
  const float* bv = (const float*)d_in[6];
  const float* wo = (const float*)d_in[7];
  const float* bo = (const float*)d_in[8];

  float* out  = (float*)d_out;                       // 4096*1024
  float* attn = out + (size_t)4096 * 1024;           // 32*2048*2048

  const size_t NE = (size_t)4096 * 1024;             // elements per activation
  float*     Qf   = (float*)d_ws;
  float*     Kf   = Qf + NE;
  float*     Vf   = Kf + NE;
  float*     OH   = Vf + NE;
  _Float16*  Q16  = (_Float16*)(OH + NE);
  _Float16*  K16  = Q16 + NE;
  _Float16*  Vt16 = K16 + NE;

  dim3 blk(256);
  dim3 gproj(4096 / 64, 1024 / 128);                 // 64 x 8
  gemm_bias_f16<<<gproj, blk, 0, stream>>>(x, wq, bq, Qf, 1024, 1024);
  gemm_bias_f16<<<gproj, blk, 0, stream>>>(x, wk, bk, Kf, 1024, 1024);
  gemm_bias_f16<<<gproj, blk, 0, stream>>>(x, wv, bv, Vf, 1024, 1024);

  rope_f16_kernel<<<8192, blk, 0, stream>>>(Qf, Q16, 0.125f);
  rope_f16_kernel<<<8192, blk, 0, stream>>>(Kf, K16, 1.0f);
  vtrans_kernel<<<16384, blk, 0, stream>>>(Vf, Vt16);

  dim3 gsc(2048 / 64, 2048 / 128, 32);
  scores_kernel<<<gsc, blk, 0, stream>>>(Q16, K16, attn);

  softmax_kernel<<<32 * 2048, blk, 0, stream>>>(attn);

  dim3 gav(2048 / 64, 1, 32);
  attnv_kernel<<<gav, blk, 0, stream>>>(attn, Vt16, OH);

  gemm_bias_f16<<<gproj, blk, 0, stream>>>(OH, wo, bo, out, 1024, 1024);
}
